// MultiHeadAttention_29291676959183
// MI455X (gfx1250) — compile-verified, matrix operand order checked by reading, not run on hardware
//
#include <hip/hip_runtime.h>
#include <hip/hip_bf16.h>

// ---------------------------------------------------------------------------
// MI455X (gfx1250) multi-head causal attention, bf16 WMMA end-to-end.
// B=4, T=2048, C=1024, H=16, D=64.
// Pipeline: cvt(x)->bf16 ; cvt+transpose W* ; QKV WMMA-GEMM ; flash-attention
// (WMMA QK^T + online softmax + WMMA PV) ; WMMA output projection (fp32 out).
// Workspace layout (bytes):
//   [0,16M)   xb      bf16 [8192,1024]
//   [16M,24M) WqT/WkT/WvT/WoT bf16 [1024,1024] each (2M apiece)
//   [24M,40M) Q bf16 [B,H,T,D]
//   [40M,56M) K bf16 [B,H,T,D]
//   [56M,72M) Vt bf16 [B,H,D,T]   (pre-transposed for PV B-fragments)
//   [72M,88M) Y bf16 [8192,1024]  (attention output, pre-projection)
// ---------------------------------------------------------------------------

#define BT   4
#define TT   2048
#define CC   1024
#define HH   16
#define DD   64
#define MM   (BT * TT)   // 8192

typedef __attribute__((ext_vector_type(16))) __bf16 v16bf;
typedef __attribute__((ext_vector_type(8)))  __bf16 v8bf;
typedef __attribute__((ext_vector_type(8)))  float  v8f;

__device__ __forceinline__ unsigned short f2bfbits(float f) {
  unsigned u; __builtin_memcpy(&u, &f, 4);
  return (unsigned short)((u + 0x7FFFu + ((u >> 16) & 1u)) >> 16);  // RNE
}
__device__ __forceinline__ __bf16 bfraw(unsigned short s) {
  __bf16 b; __builtin_memcpy(&b, &s, 2); return b;
}

// Load a 16x32 bf16 WMMA fragment (A or B) where the lane's 32 K-values are
// contiguous in memory. p points at element (row=l15, k=0) of the fragment.
// ISA layout: elems 0-7 -> K = grp*8 + 0..7 ; elems 8-15 -> K = 16 + grp*8 + 0..7
__device__ __forceinline__ v16bf ldfrag(const unsigned short* p, int grp) {
  union { uint4 u; v8bf b; } lo, hi;
  lo.u = *reinterpret_cast<const uint4*>(p + (grp << 3));
  hi.u = *reinterpret_cast<const uint4*>(p + 16 + (grp << 3));
  v16bf r;
#pragma unroll
  for (int i = 0; i < 8; ++i) { r[i] = lo.b[i]; r[i + 8] = hi.b[i]; }
  return r;
}

__device__ __forceinline__ v8f wmma_bf16(v16bf a, v16bf b, v8f c) {
  return __builtin_amdgcn_wmma_f32_16x16x32_bf16(
      /*neg_a=*/false, a, /*neg_b=*/false, b,
      /*c_mod=*/(short)0, c, /*reuse_a=*/false, /*reuse_b=*/false);
}

// ---------------------------------------------------------------------------
// fp32 -> bf16 elementwise
__global__ __launch_bounds__(256) void cvt_bf16_k(const float* __restrict__ in,
                                                  unsigned short* __restrict__ out,
                                                  long n) {
  long i = (long)blockIdx.x * 256 + threadIdx.x;
  if (i < n) out[i] = f2bfbits(in[i]);
}

// fp32 W[K,N] -> bf16 Wt[N,K]  (so B-fragments become contiguous loads)
__global__ __launch_bounds__(256) void cvt_wt_k(const float* __restrict__ W,
                                                unsigned short* __restrict__ Wt,
                                                int Kdim, int Ndim) {
  long i = (long)blockIdx.x * 256 + threadIdx.x;
  if (i < (long)Kdim * Ndim) {
    long n = i / Kdim, k = i - n * Kdim;
    Wt[i] = f2bfbits(W[k * Ndim + n]);
  }
}

// ---------------------------------------------------------------------------
// C = A[M,K] * Bt[N,K]^T + bias, bf16 inputs, fp32 accumulate via WMMA.
// Block = 256 threads = 8 waves; wave computes a 32x64 tile (2x4 16x16 frags).
// Block tile 64x256. mode: 0 -> bf16 store [B,H,T,D]; 1 -> bf16 store
// [B,H,D,T] (transposed V); 2 -> fp32 store row-major [M,N].
__global__ __launch_bounds__(256) void gemm_bf16_k(
    const unsigned short* __restrict__ A, const unsigned short* __restrict__ Bt,
    const float* __restrict__ bias, void* __restrict__ Out,
    int Mdim, int Ndim, int Kdim, int mode) {
  const int lane = threadIdx.x & 31, wave = threadIdx.x >> 5;
  const int l15 = lane & 15, grp = lane >> 4;
  const int wm = wave >> 2, wn = wave & 3;
  const long row0 = (long)blockIdx.y * 64 + wm * 32;
  const long col0 = (long)blockIdx.x * 256 + wn * 64;

  v8f acc[2][4];
#pragma unroll
  for (int i = 0; i < 2; ++i)
#pragma unroll
    for (int j = 0; j < 4; ++j) acc[i][j] = (v8f){0.f,0.f,0.f,0.f,0.f,0.f,0.f,0.f};

  for (long k0 = 0; k0 < Kdim; k0 += 32) {
    // Unconditional speculative prefetch of the next K-chunk (gfx1250
    // global_prefetch_b8 is dropped on fault; over-read stays inside the
    // workspace). Unguarded so the hot loop has no extra branch.
    __builtin_prefetch(A + (row0 + l15) * Kdim + k0 + 32, 0, 1);
    __builtin_prefetch(Bt + (col0 + l15) * Kdim + k0 + 32, 0, 1);

    v16bf af[2], bfr[4];
#pragma unroll
    for (int i = 0; i < 2; ++i)
      af[i] = ldfrag(A + (row0 + i * 16 + l15) * Kdim + k0, grp);
#pragma unroll
    for (int j = 0; j < 4; ++j)
      bfr[j] = ldfrag(Bt + (col0 + j * 16 + l15) * Kdim + k0, grp);
#pragma unroll
    for (int i = 0; i < 2; ++i)
#pragma unroll
      for (int j = 0; j < 4; ++j) acc[i][j] = wmma_bf16(af[i], bfr[j], acc[i][j]);
  }

  // Epilogue. C/D layout: element e of v8f -> M = e + 8*grp, N = l15 (per tile)
#pragma unroll
  for (int j = 0; j < 4; ++j) {
    const long n = col0 + j * 16 + l15;
    const float bv = bias[n];
    const long h = n >> 6, d = n & 63;  // n = h*D + d
#pragma unroll
    for (int i = 0; i < 2; ++i) {
#pragma unroll
      for (int e = 0; e < 8; ++e) {
        const long m = row0 + i * 16 + 8 * grp + e;
        const float v = acc[i][j][e] + bv;
        if (mode == 2) {
          ((float*)Out)[m * Ndim + n] = v;
        } else {
          const long b = m >> 11, t = m & 2047;  // m = b*T + t
          long addr;
          if (mode == 0) addr = (((b * HH + h) * TT + t) * DD + d);
          else           addr = (((b * HH + h) * DD + d) * TT + t);
          ((unsigned short*)Out)[addr] = f2bfbits(v);
        }
      }
    }
  }
}

// ---------------------------------------------------------------------------
// Flash attention (causal). grid = (B*H, T/128); block = 256 (8 waves);
// each wave owns a 16-query tile, loops key blocks of 32.
__global__ __launch_bounds__(256) void flash_attn_k(
    const unsigned short* __restrict__ Q, const unsigned short* __restrict__ Km,
    const unsigned short* __restrict__ Vt, unsigned short* __restrict__ Y) {
  __shared__ float plds[8][16 * 32];  // per-wave P staging (16KB total)

  const int lane = threadIdx.x & 31, wave = threadIdx.x >> 5;
  const int l15 = lane & 15, grp = lane >> 4;
  const int bh = blockIdx.x;            // 0..63
  const int b = bh >> 4, h = bh & 15;   // H = 16
  const int q0 = blockIdx.y * 128 + wave * 16;
  float* pl = plds[wave];

  // Q fragments (16 rows x D=64 -> two 16x32 A-frags), register resident
  const long qbase = ((long)bh * TT + q0 + l15) * DD;
  v16bf qf[2];
  qf[0] = ldfrag(Q + qbase, grp);
  qf[1] = ldfrag(Q + qbase + 32, grp);

  v8f o[4];
#pragma unroll
  for (int d = 0; d < 4; ++d) o[d] = (v8f){0.f,0.f,0.f,0.f,0.f,0.f,0.f,0.f};
  float mrow[8], lrow[8];
#pragma unroll
  for (int e = 0; e < 8; ++e) { mrow[e] = -1e30f; lrow[e] = 0.f; }

  const float scale = 0.125f;  // 1/sqrt(64)
  const int nkb = (q0 + 16 + 31) >> 5;  // causal bound, wave-uniform

  for (int kb = 0; kb < nkb; ++kb) {
    const int kt = kb * 32;
    // S = Q K^T for a 16x32 key block (two 16x16 N-tiles)
    v8f s0 = (v8f){0.f,0.f,0.f,0.f,0.f,0.f,0.f,0.f};
    v8f s1 = (v8f){0.f,0.f,0.f,0.f,0.f,0.f,0.f,0.f};
    {
      const unsigned short* kp0 = Km + ((long)bh * TT + kt + l15) * DD;
      const unsigned short* kp1 = Km + ((long)bh * TT + kt + 16 + l15) * DD;
      s0 = wmma_bf16(qf[0], ldfrag(kp0, grp), s0);
      s0 = wmma_bf16(qf[1], ldfrag(kp0 + 32, grp), s0);
      s1 = wmma_bf16(qf[0], ldfrag(kp1, grp), s1);
      s1 = wmma_bf16(qf[1], ldfrag(kp1 + 32, grp), s1);
    }
    // scale + causal mask + online softmax. Row M = e + 8*grp lives in one
    // 16-lane half; reduce across lanes with xor masks 1,2,4,8 (wave32).
    float p0[8], p1[8], alpha[8];
#pragma unroll
    for (int e = 0; e < 8; ++e) {
      const int q  = q0 + 8 * grp + e;
      const float a = (kt + l15      <= q) ? s0[e] * scale : -1e30f;
      const float c = (kt + 16 + l15 <= q) ? s1[e] * scale : -1e30f;
      float mx = fmaxf(a, c);
      mx = fmaxf(mx, __shfl_xor(mx, 1));
      mx = fmaxf(mx, __shfl_xor(mx, 2));
      mx = fmaxf(mx, __shfl_xor(mx, 4));
      mx = fmaxf(mx, __shfl_xor(mx, 8));
      const float mn = fmaxf(mrow[e], mx);
      alpha[e] = __expf(mrow[e] - mn);
      mrow[e] = mn;
      p0[e] = __expf(a - mn);
      p1[e] = __expf(c - mn);
      float rs = p0[e] + p1[e];
      rs += __shfl_xor(rs, 1);
      rs += __shfl_xor(rs, 2);
      rs += __shfl_xor(rs, 4);
      rs += __shfl_xor(rs, 8);
      lrow[e] = lrow[e] * alpha[e] + rs;
    }
#pragma unroll
    for (int d = 0; d < 4; ++d)
#pragma unroll
      for (int e = 0; e < 8; ++e) o[d][e] *= alpha[e];

    // Relayout P (C-layout) -> A-fragment layout through per-wave LDS.
#pragma unroll
    for (int e = 0; e < 8; ++e) {
      const int r = 8 * grp + e;
      pl[r * 32 + l15]      = p0[e];
      pl[r * 32 + 16 + l15] = p1[e];
    }
    asm volatile("s_wait_dscnt 0x0" ::: "memory");  // cross-lane LDS WAR->RAW
    const float* srcp = pl + l15 * 32;
    float4 a0 = *(const float4*)(srcp + grp * 8);
    float4 a1 = *(const float4*)(srcp + grp * 8 + 4);
    float4 a2 = *(const float4*)(srcp + 16 + grp * 8);
    float4 a3 = *(const float4*)(srcp + 16 + grp * 8 + 4);
    asm volatile("s_wait_dscnt 0x0" ::: "memory");  // fence before next-iter stores
    float pv[16] = {a0.x,a0.y,a0.z,a0.w, a1.x,a1.y,a1.z,a1.w,
                    a2.x,a2.y,a2.z,a2.w, a3.x,a3.y,a3.z,a3.w};
    v16bf pf;
#pragma unroll
    for (int i = 0; i < 16; ++i) pf[i] = bfraw(f2bfbits(pv[i]));

    // O += P * V  (V pre-transposed [B,H,D,T] -> contiguous B-frag loads)
#pragma unroll
    for (int d = 0; d < 4; ++d) {
      const unsigned short* vp = Vt + ((long)bh * DD + d * 16 + l15) * TT + kt;
      o[d] = wmma_bf16(pf, ldfrag(vp, grp), o[d]);
    }
  }

  // Normalize and store Y[b, q, h*D + d] as bf16
  float inv[8];
#pragma unroll
  for (int e = 0; e < 8; ++e) inv[e] = 1.f / lrow[e];
#pragma unroll
  for (int d = 0; d < 4; ++d)
#pragma unroll
    for (int e = 0; e < 8; ++e) {
      const long q = q0 + 8 * grp + e;
      const long addr = ((long)b * TT + q) * CC + h * DD + d * 16 + l15;
      Y[addr] = f2bfbits(o[d][e] * inv[e]);
    }
}

// ---------------------------------------------------------------------------
extern "C" void kernel_launch(void* const* d_in, const int* in_sizes, int n_in,
                              void* d_out, int out_size, void* d_ws, size_t ws_size,
                              hipStream_t stream) {
  (void)in_sizes; (void)n_in; (void)out_size; (void)ws_size;
  const float* x  = (const float*)d_in[0];
  const float* Wq = (const float*)d_in[1];
  const float* bq = (const float*)d_in[2];
  const float* Wk = (const float*)d_in[3];
  const float* bk = (const float*)d_in[4];
  const float* Wv = (const float*)d_in[5];
  const float* bv = (const float*)d_in[6];
  const float* Wo = (const float*)d_in[7];
  const float* bo = (const float*)d_in[8];

  char* base = (char*)d_ws;
  const size_t MiB = 1ull << 20;
  unsigned short* xb  = (unsigned short*)(base);
  unsigned short* wqt = (unsigned short*)(base + 16 * MiB);
  unsigned short* wkt = (unsigned short*)(base + 18 * MiB);
  unsigned short* wvt = (unsigned short*)(base + 20 * MiB);
  unsigned short* wot = (unsigned short*)(base + 22 * MiB);
  unsigned short* Qb  = (unsigned short*)(base + 24 * MiB);
  unsigned short* Kb  = (unsigned short*)(base + 40 * MiB);
  unsigned short* Vtb = (unsigned short*)(base + 56 * MiB);
  unsigned short* Yb  = (unsigned short*)(base + 72 * MiB);

  const long nx = (long)MM * CC;          // 8388608
  cvt_bf16_k<<<(unsigned)(nx / 256), 256, 0, stream>>>(x, xb, nx);
  const unsigned nw = (CC * CC) / 256;    // 4096
  cvt_wt_k<<<nw, 256, 0, stream>>>(Wq, wqt, CC, CC);
  cvt_wt_k<<<nw, 256, 0, stream>>>(Wk, wkt, CC, CC);
  cvt_wt_k<<<nw, 256, 0, stream>>>(Wv, wvt, CC, CC);
  cvt_wt_k<<<nw, 256, 0, stream>>>(Wo, wot, CC, CC);

  dim3 gg(CC / 256, MM / 64);  // (4, 128)
  gemm_bf16_k<<<gg, 256, 0, stream>>>(xb, wqt, bq, Qb,  MM, CC, CC, 0);
  gemm_bf16_k<<<gg, 256, 0, stream>>>(xb, wkt, bk, Kb,  MM, CC, CC, 0);
  gemm_bf16_k<<<gg, 256, 0, stream>>>(xb, wvt, bv, Vtb, MM, CC, CC, 1);

  flash_attn_k<<<dim3(BT * HH, TT / 128), 256, 0, stream>>>(Qb, Kb, Vtb, Yb);

  gemm_bf16_k<<<gg, 256, 0, stream>>>(Yb, wot, bo, d_out, MM, CC, CC, 2);
}
